// ODE_RNN_2989297238709
// MI455X (gfx1250) — matrix operand order repaired and checked by reading, not compiled, not run on hardware
//
#include <hip/hip_runtime.h>
#include <hip/hip_bf16.h>

// ---------------------------------------------------------------------------
// ODE-RNN fused persistent kernel for MI455X (gfx1250, wave32, WMMA).
// Batch rows are independent across the T=128 recurrence -> each workgroup
// owns 16 batch rows (one WMMA M-tile) and iterates all timesteps with
// h / hstd resident in LDS. Weights are packed once to bf16 fragment-major
// layout (matches V_WMMA B-operand lane layout) and streamed from L2 every
// step (weights = 6.25 MB bf16 << 192 MB L2; intensity 16 flop/B from L2).
// ---------------------------------------------------------------------------

typedef __bf16 bf16_t;
typedef __attribute__((ext_vector_type(16))) __bf16 v16bf;
typedef __attribute__((ext_vector_type(8)))  __bf16 v8bf;
typedef __attribute__((ext_vector_type(8)))  float  v8f;

#define B_SZ   1024
#define T_SZ   128
#define D_SZ   256
#define H_SZ   512
#define HID_SZ 512
#define C_SZ   1280   // 2H + D

// packed-weight region offsets (in bf16 elements) inside d_ws
#define W1P_OFF  0
#define W2P_OFF  (512*512)
#define WZRP_OFF (2*512*512)
#define WNP_OFF  (2*512*512 + 1280*1024)
#define PACK_TOTAL (2*512*512 + 2*1280*1024)   // 3,145,728 bf16 = 6 MB

__device__ __forceinline__ float sigm(float v) { return 1.0f / (1.0f + __expf(-v)); }

// ---------------------------------------------------------------------------
// Weight pack: f32 row-major [K][N] -> bf16 fragment-major.
// Fragment (kk, nn) covers K rows [kk*32, kk*32+32), N cols [nn*16, nn*16+16).
// Within a fragment: 32 lanes x 16 halves, lane<16 -> (N=lane, K=kk*32+{0..7,16..23}),
// lane>=16 -> (N=lane-16, K=kk*32+{8..15,24..31}).  srcB!=null splits N at nsplit
// (used to fuse Wz|Wr into one 1024-wide operand).
// ---------------------------------------------------------------------------
__global__ void pack_weights_k(const float* __restrict__ srcA,
                               const float* __restrict__ srcB,
                               int K, int N, int nsplit,
                               bf16_t* __restrict__ dst, int total) {
  int idx = blockIdx.x * blockDim.x + threadIdx.x;
  if (idx >= total) return;
  int i    = idx & 15;
  int lane = (idx >> 4) & 31;
  int frag = idx >> 9;
  int ntn  = N >> 4;
  int kk   = frag / ntn;
  int nn   = frag - kk * ntn;
  int k = (kk << 5) + ((lane >> 4) << 3) + (i & 7) + ((i >> 3) << 4);
  int n = (nn << 4) + (lane & 15);
  float v;
  if (n < nsplit) v = srcA[(size_t)k * nsplit + n];
  else            v = srcB[(size_t)k * (N - nsplit) + (n - nsplit)];
  dst[idx] = (bf16_t)v;
}

// ---------------------------------------------------------------------------
// 16-row GEMM inner loop: C[16,16*NT] += A[16,K] (LDS bf16) * Bpacked.
// A fragment per ISA 16-bit A layout: lane<16 holds M=lane, K {0..7,16..23};
// lane>=16 holds M=lane-16, K {8..15,24..31}  -> two ds_load_b128 per chunk.
// B fragment: one contiguous 32B/lane read from packed L2-resident weights.
//
// zofs is an opaque zero (volatile asm): it blocks LICM/CSE of the
// t-invariant G1/G2 weight loads (round-1 asm showed them hoisted out of the
// timestep loop and spilled to scratch) WITHOUT laundering the pointer
// itself, so provenance from the kernarg global pointer survives and the
// backend emits global_load_b128 / global_prefetch_b8 instead of FLAT ops
// (round-2 asm showed flat_load_b128, which double-counts into DScnt and
// serializes against the LDS A-fragment pipeline via s_wait_loadcnt_dscnt).
// ---------------------------------------------------------------------------
template <int NT>
__device__ __forceinline__ void gemm16(const bf16_t* __restrict__ A, int K,
                                       const bf16_t* __restrict__ Bp, int ntn,
                                       int tile0, int lane, v8f (&acc)[NT]) {
  size_t zofs = 0;
  asm volatile("" : "+s"(zofs));           // opaque 0, provenance-preserving
  const v16bf* Bv = (const v16bf*)(Bp + zofs);
  const int m  = lane & 15;
  const int kh = (lane & 16) ? 8 : 0;
  const bf16_t* arow = A + m * K + kh;
  const int chunks = K >> 5;
  for (int kk = 0; kk < chunks; ++kk) {
    const bf16_t* ap = arow + (kk << 5);
    v8bf lo = *(const v8bf*)ap;
    v8bf hi = *(const v8bf*)(ap + 16);
    v16bf av = __builtin_shufflevector(lo, hi, 0,1,2,3,4,5,6,7,
                                               8,9,10,11,12,13,14,15);
    if (kk + 1 < chunks) {
      const v16bf* pf = Bv + (((size_t)(kk + 1) * ntn + tile0) << 5) + lane;
      __builtin_prefetch((const void*)pf, 0, 3);   // next K-chunk into near cache
    }
#pragma unroll
    for (int i = 0; i < NT; ++i) {
      v16bf bv = Bv[(((size_t)kk * ntn + tile0 + i) << 5) + lane];
      acc[i] = __builtin_amdgcn_wmma_f32_16x16x32_bf16(
          false, av, false, bv, (short)0, acc[i], false, false);
    }
  }
}

// ---------------------------------------------------------------------------
// Persistent per-batch-tile recurrence kernel. 512 threads = 16 waves.
// ---------------------------------------------------------------------------
__global__ __launch_bounds__(512, 1)
void odernn_fused(const float* __restrict__ x,  const float* __restrict__ xt,
                  const float* __restrict__ b1, const float* __restrict__ b2,
                  const float* __restrict__ bz, const float* __restrict__ br,
                  const float* __restrict__ bn,
                  const bf16_t* __restrict__ wpack,
                  float* __restrict__ out) {
  extern __shared__ unsigned char smem[];
  float*  hf    = (float*)smem;                 // [16][512] f32 state h
  float*  stdf  = hf    + 16 * H_SZ;            // [16][512] f32 state hstd
  float*  hodef = stdf  + 16 * H_SZ;            // [16][512] f32 h_ode
  float*  zbuf  = hodef + 16 * H_SZ;            // [16][512] f32 update gate
  bf16_t* hb    = (bf16_t*)(zbuf + 16 * H_SZ);  // [16][512] bf16 h (G1 A)
  bf16_t* a1b   = hb   + 16 * H_SZ;             // [16][512] bf16 tanh-act (G2 A)
  bf16_t* catA  = a1b  + 16 * H_SZ;             // [16][1280] bf16 [h_ode|hstd|x]
  bf16_t* catR  = catA + 16 * C_SZ;             // [16][1280] bf16 [h_ode*r|hstd*r|x]

  const bf16_t* W1p  = wpack + W1P_OFF;
  const bf16_t* W2p  = wpack + W2P_OFF;
  const bf16_t* WZRp = wpack + WZRP_OFF;
  const bf16_t* WNp  = wpack + WNP_OFF;

  const int tid  = threadIdx.x;
  const int lane = tid & 31;
  const int wave = tid >> 5;
  const int b0   = blockIdx.x * 16;
  const int nC   = lane & 15;            // C/D layout: N = lane&15
  const int mOff = (lane >> 4) << 3;     // C/D layout: M = r + 8*(lane>=16)

  // h0 = 0, std0 = 0; catA std/x columns zeroed (std cols refreshed by G4)
  for (int i = tid; i < 16 * H_SZ; i += 512) {
    hf[i] = 0.f; stdf[i] = 0.f; hb[i] = (bf16_t)0.f;
  }
  for (int i = tid; i < 16 * C_SZ; i += 512) catA[i] = (bf16_t)0.f;
  __syncthreads();

  for (int t = 0; t < T_SZ; ++t) {
    const float dt = (t == 0) ? 0.01f : (xt[t] - xt[t - 1]);

    // stage x_t (bf16) into both concat operands
    for (int i = tid; i < 16 * D_SZ; i += 512) {
      int m = i >> 8, d = i & (D_SZ - 1);
      bf16_t xv = (bf16_t)x[((size_t)(b0 + m) * T_SZ + t) * D_SZ + d];
      catA[m * C_SZ + 2 * H_SZ + d] = xv;
      catR[m * C_SZ + 2 * H_SZ + d] = xv;
    }
    __syncthreads();

    // G1: a1 = tanh(h @ W1 + b1)          [16,512] x [512,512]
    {
      v8f acc[2] = {};
      gemm16<2>(hb, H_SZ, W1p, H_SZ / 16, wave * 2, lane, acc);
#pragma unroll
      for (int i = 0; i < 2; ++i) {
        int n = (wave * 2 + i) * 16 + nC;
#pragma unroll
        for (int r = 0; r < 8; ++r) {
          int m = r + mOff;
          a1b[m * H_SZ + n] = (bf16_t)tanhf(acc[i][r] + b1[n]);
        }
      }
    }
    __syncthreads();

    // G2: h_ode = h + dt * (a1 @ W2 + b2)
    {
      v8f acc[2] = {};
      gemm16<2>(a1b, H_SZ, W2p, H_SZ / 16, wave * 2, lane, acc);
#pragma unroll
      for (int i = 0; i < 2; ++i) {
        int n = (wave * 2 + i) * 16 + nC;
#pragma unroll
        for (int r = 0; r < 8; ++r) {
          int m = r + mOff;
          float ho = hf[m * H_SZ + n] + dt * (acc[i][r] + b2[n]);
          hodef[m * H_SZ + n] = ho;
          catA[m * C_SZ + n]  = (bf16_t)ho;
        }
      }
    }
    __syncthreads();

    // G3: [z|r] = sigmoid(cat @ [Wz|Wr] + [bz|br])   [16,1280] x [1280,1024]
    {
      v8f acc[4] = {};
      gemm16<4>(catA, C_SZ, WZRp, 64, wave * 4, lane, acc);
#pragma unroll
      for (int i = 0; i < 4; ++i) {
        int tile = wave * 4 + i;
#pragma unroll
        for (int r = 0; r < 8; ++r) {
          int m = r + mOff;
          if (tile < 32) {                       // z columns
            int n = tile * 16 + nC;
            zbuf[m * H_SZ + n] = sigm(acc[i][r] + bz[n]);
          } else {                               // r columns -> build cat_r
            int n = (tile - 32) * 16 + nC;
            float rv = sigm(acc[i][r] + br[n]);
            catR[m * C_SZ + n]        = (bf16_t)(hodef[m * H_SZ + n] * rv);
            catR[m * C_SZ + H_SZ + n] = (bf16_t)(stdf [m * H_SZ + n] * rv);
          }
        }
      }
    }
    __syncthreads();

    // G4: [n_mean|n_std] = cat_r @ Wn + bn ; GRU blend -> next-step state
    {
      v8f acc[4] = {};
      gemm16<4>(catR, C_SZ, WNp, 64, wave * 4, lane, acc);
#pragma unroll
      for (int i = 0; i < 4; ++i) {
        int tile = wave * 4 + i;
#pragma unroll
        for (int r = 0; r < 8; ++r) {
          int m = r + mOff;
          if (tile < 32) {                       // mean half
            int n = tile * 16 + nC;
            float nm = acc[i][r] + bn[n];
            float zv = zbuf[m * H_SZ + n];
            float hn = (1.f - zv) * nm + zv * hodef[m * H_SZ + n];
            hf[m * H_SZ + n] = hn;
            hb[m * H_SZ + n] = (bf16_t)hn;       // G1 operand, next step
          } else {                               // std half
            int n = (tile - 32) * 16 + nC;
            float ns = fabsf(acc[i][r] + bn[H_SZ + n]);
            float zv = zbuf[m * H_SZ + n];
            float sn = fabsf((1.f - zv) * ns + zv * stdf[m * H_SZ + n]);
            stdf[m * H_SZ + n] = sn;
            catA[m * C_SZ + H_SZ + n] = (bf16_t)sn;  // G3 operand, next step
          }
        }
      }
    }
    __syncthreads();
  }

  // emit (hT, stdT) concatenated flat, f32
  for (int i = tid; i < 16 * H_SZ; i += 512) {
    int m = i >> 9, n = i & (H_SZ - 1);
    out[(size_t)(b0 + m) * H_SZ + n] = hf[i];
    out[(size_t)B_SZ * H_SZ + (size_t)(b0 + m) * H_SZ + n] = stdf[i];
  }
}

#define SMEM_BYTES (4*(16*H_SZ)*4 + 2*(16*H_SZ)*2 + 2*(16*C_SZ)*2)  // 245760 B

extern "C" void kernel_launch(void* const* d_in, const int* in_sizes, int n_in,
                              void* d_out, int out_size, void* d_ws, size_t ws_size,
                              hipStream_t stream) {
  const float* x  = (const float*)d_in[0];
  const float* xt = (const float*)d_in[1];
  const float* W1 = (const float*)d_in[2];
  const float* b1 = (const float*)d_in[3];
  const float* W2 = (const float*)d_in[4];
  const float* b2 = (const float*)d_in[5];
  const float* Wz = (const float*)d_in[6];
  const float* bz = (const float*)d_in[7];
  const float* Wr = (const float*)d_in[8];
  const float* br = (const float*)d_in[9];
  const float* Wn = (const float*)d_in[10];
  const float* bn = (const float*)d_in[11];
  bf16_t* ws = (bf16_t*)d_ws;
  float* out = (float*)d_out;

  // 1) repack weights to bf16 fragment-major (runs every call: deterministic)
  {
    int tot;
    tot = 512 * 512;
    pack_weights_k<<<(tot + 255) / 256, 256, 0, stream>>>(W1, nullptr, 512, 512, 512, ws + W1P_OFF, tot);
    pack_weights_k<<<(tot + 255) / 256, 256, 0, stream>>>(W2, nullptr, 512, 512, 512, ws + W2P_OFF, tot);
    tot = 1280 * 1024;
    pack_weights_k<<<(tot + 255) / 256, 256, 0, stream>>>(Wz, Wr, 1280, 1024, 512, ws + WZRP_OFF, tot);
    pack_weights_k<<<(tot + 255) / 256, 256, 0, stream>>>(Wn, nullptr, 1280, 1024, 1024, ws + WNP_OFF, tot);
  }

  // 2) persistent recurrence: 64 batch tiles x 16 waves, 240 KB LDS each.
  //    Attribute set is idempotent and runs every call (deterministic).
  hipFuncSetAttribute((const void*)odernn_fused,
                      hipFuncAttributeMaxDynamicSharedMemorySize, SMEM_BYTES);
  odernn_fused<<<B_SZ / 16, 512, SMEM_BYTES, stream>>>(
      x, xt, b1, b2, bz, br, bn, ws, out);
}
